// MLPAttention_35029753266668
// MI455X (gfx1250) — matrix operand order repaired and checked by reading, not compiled, not run on hardware
//
#include <hip/hip_runtime.h>
#include <hip/hip_bf16.h>
#include <math.h>
#include <stdint.h>

typedef __attribute__((ext_vector_type(16))) __bf16 v16bf;
typedef __attribute__((ext_vector_type(8)))  float  v8f;

#if __has_builtin(__builtin_amdgcn_tanhf)
#define FAST_TANH(x) __builtin_amdgcn_tanhf(x)
#else
#define FAST_TANH(x) tanhf(x)
#endif

#define NEGV   (-1e6f)
#define B_SZ   16
#define LQ     128
#define LK     256
#define DIM    512   // Dq = Dk = Dv
#define UDIM   256
#define QPS    260   // qp LDS row stride (floats); 260*4 % 16 == 0 for b128 async stores
#define KPS    260   // kp LDS row stride
#define VTS    520   // value LDS row stride
#define SRS    257   // scores LDS row stride

// ---- CDNA5 async global -> LDS copy (ASYNCcnt-tracked, bypasses VGPRs) -----
__device__ __forceinline__ uint32_t lds_off(const void* p) {
  // addrspace(3) offset == low 32 bits of the generic (flat) LDS address
  return (uint32_t)(uintptr_t)p;
}
__device__ __forceinline__ void async_b128(const float* gsrc, uint32_t lds_byte_off) {
  asm volatile("global_load_async_to_lds_b128 %0, %1, off"
               :: "v"(lds_byte_off), "v"(gsrc)
               : "memory");
}
__device__ __forceinline__ void wait_async() {
  asm volatile("s_wait_asynccnt 0x0" ::: "memory");
}

// ---------------------------------------------------------------------------
// Kernel A: C[M x 256] = A[M x 512] * W[512 x 256], bf16 WMMA, fp32 accum.
// Block tile 128(M) x 64(N), 8 waves in a 4x2 grid, each wave a 32x32 subtile
// (2x2 WMMAs reusing 2 A-frags and 2 B-frags).
// ---------------------------------------------------------------------------
__global__ __launch_bounds__(256) void proj_gemm_kernel(
    const float* __restrict__ A, const float* __restrict__ W,
    float* __restrict__ C) {
  __shared__ __align__(16) __bf16 As[128][34];  // [m][k]; 68B stride, conflict-free
  __shared__ __align__(16) __bf16 Bs[64][34];   // transposed: [n][k]

  const int t     = threadIdx.x;
  const int lane  = t & 31;
  const int wave  = t >> 5;
  const int m_sub = wave & 3;     // 0..3 -> 32-row band
  const int n_sub = wave >> 2;    // 0..1 -> 32-col band
  const int m0    = blockIdx.x * 128;
  const int n0    = blockIdx.y * 64;

  v8f acc00 = {}, acc01 = {}, acc10 = {}, acc11 = {};

  for (int k0 = 0; k0 < DIM; k0 += 32) {
    // Stage A tile 128x32 (each thread: 16 consecutive floats -> bf16)
    {
      const int row = t >> 1;
      const int cb  = (t & 1) * 16;
      const float* src = A + (size_t)(m0 + row) * DIM + k0 + cb;
#pragma unroll
      for (int j = 0; j < 16; ++j) As[row][cb + j] = (__bf16)src[j];
    }
    // Stage W tile 32x64, transposed to [n][k]
    {
      const int kr = t >> 3;
      const int nb = (t & 7) * 8;
      const float* src = W + (size_t)(k0 + kr) * UDIM + n0 + nb;
#pragma unroll
      for (int j = 0; j < 8; ++j) Bs[nb + j][kr] = (__bf16)src[j];
    }
    __syncthreads();

    // A/B fragments in contiguous k order (k = 16*(lane/16) + e), same order for A and B
    v16bf a0, a1, b0, b1;
#pragma unroll
    for (int e = 0; e < 16; ++e) {
      const int Ka = ((lane >= 16) ? 16 : 0) + e;
      a0[e] = As[m_sub * 32 + (lane & 15)][Ka];
      a1[e] = As[m_sub * 32 + 16 + (lane & 15)][Ka];
      const int Kb = ((lane >= 16) ? 16 : 0) + e;
      b0[e] = Bs[n_sub * 32 + (lane & 15)][Kb];
      b1[e] = Bs[n_sub * 32 + 16 + (lane & 15)][Kb];
    }
    acc00 = __builtin_amdgcn_wmma_f32_16x16x32_bf16(false, a0, false, b0, (short)0, acc00, false, false);
    acc01 = __builtin_amdgcn_wmma_f32_16x16x32_bf16(false, a0, false, b1, (short)0, acc01, false, false);
    acc10 = __builtin_amdgcn_wmma_f32_16x16x32_bf16(false, a1, false, b0, (short)0, acc10, false, false);
    acc11 = __builtin_amdgcn_wmma_f32_16x16x32_bf16(false, a1, false, b1, (short)0, acc11, false, false);
    __syncthreads();
  }

  // Store 4 subtiles: lane col = L%16; VGPR i row = i + (L>=16 ? 8 : 0)
  const int cbase = n0 + n_sub * 32 + (lane & 15);
  const int rbase = m0 + m_sub * 32 + ((lane >= 16) ? 8 : 0);
  const v8f* accs[4] = {&acc00, &acc01, &acc10, &acc11};
#pragma unroll
  for (int s = 0; s < 4; ++s) {
    const int cn = cbase + (s & 1) * 16;
    const int rm = rbase + (s >> 1) * 16;
#pragma unroll
    for (int i = 0; i < 8; ++i)
      C[(size_t)(rm + i) * UDIM + cn] = (*accs[s])[i];
  }
}

// ---------------------------------------------------------------------------
// Kernel B: per (batch, 16-query tile): tanh scores -> masked softmax -> AV.
// 512 threads (16 waves). Async global->LDS staging, ping-pong pipelined so
// the tile-(i+1) DMA hides behind the tile-i tanh loop.
// ---------------------------------------------------------------------------
__global__ __launch_bounds__(512) void attn_fused_kernel(
    const float* __restrict__ qp, const float* __restrict__ kp,
    const float* __restrict__ value, const int* __restrict__ valid_len,
    const float* __restrict__ vw_g, float* __restrict__ out) {
  __shared__ __align__(16) float qs[16 * QPS];
  __shared__ __align__(16) float sc[16 * SRS];
  __shared__ __align__(16) float vw[UDIM];
  __shared__ __align__(16) union SMemU {
    float kp2[2][32 * KPS];  // phase 1: ping-pong key-projection tiles (66.5 KB)
    float vt[32 * VTS];      // phase 2: value tile (same 66.5 KB footprint)
  } smem;

  const int b    = blockIdx.y;
  const int qt   = blockIdx.x;          // 0..7
  const int t    = threadIdx.x;
  const int lane = t & 31;
  const int wave = t >> 5;              // 0..15
  const int vlen = valid_len[b];

  auto stage_kp = [&](int kt, int buf) {
    const float* kpb = kp + (size_t)(b * LK + kt) * UDIM;   // 32x256 contiguous
#pragma unroll
    for (int c = t; c < 2048; c += 512) {                   // 32 rows * 64 b128 chunks
      const int row = c >> 6, col = (c & 63) * 4;
      async_b128(kpb + row * UDIM + col, lds_off(&smem.kp2[buf][row * KPS + col]));
    }
  };
  auto stage_vt = [&](int kt) {
    const float* vb = value + (size_t)(b * LK + kt) * DIM;  // 32x512 contiguous
#pragma unroll
    for (int c = t; c < 4096; c += 512) {                   // 32 rows * 128 b128 chunks
      const int row = c >> 7, col = (c & 127) * 4;
      async_b128(vb + row * DIM + col, lds_off(&smem.vt[row * VTS + col]));
    }
  };

  // Async-stage q-projection tile (16x256 contiguous -> padded rows), v_w,
  // and the first kp tile.
  {
    const float* qpb = qp + (size_t)(b * LQ + qt * 16) * UDIM;
#pragma unroll
    for (int c = t; c < 1024; c += 512) {                   // 16 rows * 64 b128 chunks
      const int row = c >> 6, col = (c & 63) * 4;
      async_b128(qpb + row * UDIM + col, lds_off(&qs[row * QPS + col]));
    }
    if (t < 64) async_b128(vw_g + t * 4, lds_off(&vw[t * 4]));
  }
  stage_kp(0, 0);

  // ---- Phase 1: additive scores (v_tanh path); 1 (q,k) pair per thread ----
  const int q  = wave;     // query row 0..15
  const int kl = lane;     // key within 32-tile
  for (int i = 0; i < 8; ++i) {
    wait_async();                      // tile i (and qs/vw on i==0) landed
    __syncthreads();
    if (i < 7) stage_kp((i + 1) * 32, (i + 1) & 1);   // DMA next tile during tanh

    float s = 0.f;
    const float* qrow = &qs[q * QPS];
    const float* krow = &smem.kp2[i & 1][kl * KPS];
#pragma unroll 4
    for (int u = 0; u < UDIM; ++u)
      s += vw[u] * FAST_TANH(qrow[u] + krow[u]);
    const int kg = i * 32 + kl;
    sc[q * SRS + kg] = (kg < vlen) ? s : NEGV;
  }
  __syncthreads();                     // scores complete; kp reads complete

  // Prefetch first value tile while softmax runs (union region now free)
  stage_vt(0);

  // ---- Masked softmax: one wave per row, wave32 shuffle reductions ----
  {
    float* r = &sc[wave * SRS];
    float vals[8];
    float mx = -3.4e38f;
#pragma unroll
    for (int j = 0; j < 8; ++j) { vals[j] = r[lane + j * 32]; mx = fmaxf(mx, vals[j]); }
#pragma unroll
    for (int off = 16; off > 0; off >>= 1) mx = fmaxf(mx, __shfl_xor(mx, off, 32));
    float sum = 0.f;
#pragma unroll
    for (int j = 0; j < 8; ++j) { vals[j] = __expf(vals[j] - mx); sum += vals[j]; }
#pragma unroll
    for (int off = 16; off > 0; off >>= 1) sum += __shfl_xor(sum, off, 32);
    const float inv = 1.0f / sum;
#pragma unroll
    for (int j = 0; j < 8; ++j) r[lane + j * 32] = vals[j] * inv;
  }
  __syncthreads();

  // ---- Phase 2: out(16x512) = attn(16x256) @ value_b(256x512), bf16 WMMA ----
  // 16 waves x 2 n-subtiles of 16 -> 512 columns.
  v8f acc0 = {}, acc1 = {};
  for (int i = 0; i < 8; ++i) {
    const int kt = i * 32;
    wait_async();
    __syncthreads();                   // tile i resident for all threads

    v16bf a;
#pragma unroll
    for (int e = 0; e < 16; ++e) {
      const int Ka = kt + ((lane >= 16) ? 16 : 0) + e;
      a[e] = (__bf16)sc[(lane & 15) * SRS + Ka];
    }
    {
      const int n0 = (wave * 2) * 16;
      v16bf bf;
#pragma unroll
      for (int e = 0; e < 16; ++e) {
        const int Kb = ((lane >= 16) ? 16 : 0) + e;
        bf[e] = (__bf16)smem.vt[Kb * VTS + n0 + (lane & 15)];
      }
      acc0 = __builtin_amdgcn_wmma_f32_16x16x32_bf16(false, a, false, bf, (short)0, acc0, false, false);
    }
    {
      const int n0 = (wave * 2 + 1) * 16;
      v16bf bf;
#pragma unroll
      for (int e = 0; e < 16; ++e) {
        const int Kb = ((lane >= 16) ? 16 : 0) + e;
        bf[e] = (__bf16)smem.vt[Kb * VTS + n0 + (lane & 15)];
      }
      acc1 = __builtin_amdgcn_wmma_f32_16x16x32_bf16(false, a, false, bf, (short)0, acc1, false, false);
    }

    if (i < 7) {
      __syncthreads();                 // all vt reads done before overwrite
      stage_vt(kt + 32);
    }
  }

  // Store output tile (16 x 512)
  float* ob = out + (size_t)(b * LQ + qt * 16) * DIM;
#pragma unroll
  for (int i = 0; i < 2; ++i) {
    const int cn = (wave * 2 + i) * 16 + (lane & 15);
    const v8f& a8 = i ? acc1 : acc0;
#pragma unroll
    for (int r8 = 0; r8 < 8; ++r8) {
      const int row = r8 + ((lane >= 16) ? 8 : 0);
      ob[(size_t)row * DIM + cn] = a8[r8];
    }
  }
}

// ---------------------------------------------------------------------------
extern "C" void kernel_launch(void* const* d_in, const int* in_sizes, int n_in,
                              void* d_out, int out_size, void* d_ws, size_t ws_size,
                              hipStream_t stream) {
  const float* query     = (const float*)d_in[0];   // (16,128,512)
  const float* key       = (const float*)d_in[1];   // (16,256,512)
  const float* value     = (const float*)d_in[2];   // (16,256,512)
  const int*   valid_len = (const int*)  d_in[3];   // (16,)
  const float* W_q       = (const float*)d_in[4];   // (512,256)
  const float* W_k       = (const float*)d_in[5];   // (512,256)
  const float* v_w       = (const float*)d_in[6];   // (256,)
  float* out = (float*)d_out;                       // (16,128,512)

  float* qp_ws = (float*)d_ws;                       // 2048 x 256 fp32 (2 MB)
  float* kp_ws = qp_ws + (size_t)(B_SZ * LQ) * UDIM; // 4096 x 256 fp32 (4 MB)

  // Projections: block tile 128(M) x 64(N)
  proj_gemm_kernel<<<dim3((B_SZ * LQ) / 128, UDIM / 64), 256, 0, stream>>>(query, W_q, qp_ws);
  proj_gemm_kernel<<<dim3((B_SZ * LK) / 128, UDIM / 64), 256, 0, stream>>>(key,   W_k, kp_ws);

  // Fused tanh-scores -> masked softmax -> AV
  attn_fused_kernel<<<dim3(LQ / 16, B_SZ), 512, 0, stream>>>(
      qp_ws, kp_ws, value, valid_len, v_w, out);
}
